// MultiHeadSelfAttention_87875030876855
// MI455X (gfx1250) — matrix operand order repaired and checked by reading, not compiled, not run on hardware
//
#include <hip/hip_runtime.h>
#include <hip/hip_bf16.h>

typedef __attribute__((ext_vector_type(16))) __bf16 v16bf;
typedef __attribute__((ext_vector_type(8)))  __bf16 v8bf;
typedef __attribute__((ext_vector_type(8)))  float  v8f;
typedef __attribute__((ext_vector_type(4)))  unsigned int tdm_u32x4;
typedef __attribute__((ext_vector_type(8)))  int          tdm_i32x8;
typedef __attribute__((ext_vector_type(4)))  int          tdm_i32x4;

#define D_MODEL 1024
#define N_HEADS 16
#define DK      64
#define SEQ     2048
#define BATCH   4
#define MROWS   (BATCH * SEQ)   // 8192

#if defined(__gfx1250__) && __has_builtin(__builtin_amdgcn_tensor_load_to_lds)
#define HAVE_TDM 1
#else
#define HAVE_TDM 0
#endif

// ---------- helpers ----------

__device__ __forceinline__ __bf16 f2bf(float f) {
  unsigned u = __builtin_bit_cast(unsigned, f);
  u += 0x7FFFu + ((u >> 16) & 1u);                 // round-to-nearest-even
  unsigned short h = (unsigned short)(u >> 16);
  return __builtin_bit_cast(__bf16, h);
}

__device__ __forceinline__ v8f wmma_bf16(v16bf a, v16bf b, v8f c) {
  // D = A(16x32 bf16) * B(32x16 bf16) + C(16x16 f32)
  return __builtin_amdgcn_wmma_f32_16x16x32_bf16(
      false, a, false, b, (short)0, c, false, false);
}

// A fragment: lane holds row m = lane&15; laneHalf=0: halves K 0..15;
// laneHalf=1: halves K 16..31.  row_k0 points at (row, K=0 of the step).
__device__ __forceinline__ v16bf load_a_frag(const __bf16* row_k0, int laneHalf) {
  v8bf lo = *reinterpret_cast<const v8bf*>(row_k0 + 16 * laneHalf);
  v8bf hi = *reinterpret_cast<const v8bf*>(row_k0 + 16 * laneHalf + 8);
  v16bf a;
#pragma unroll
  for (int i = 0; i < 8; ++i) { a[i] = lo[i]; a[i + 8] = hi[i]; }
  return a;
}

// B fragment: lane holds col n = lane&15; halves = K (16*laneHalf+0..15),
// contiguous in memory (column-major B).
__device__ __forceinline__ v16bf load_b_frag(const __bf16* col_k0, int laneHalf) {
  return *reinterpret_cast<const v16bf*>(col_k0 + 16 * laneHalf);
}

#if HAVE_TDM
// 2-D Tensor-DMA load into LDS: tile_d0 contiguous bf16 elements per row,
// tile_d1 rows, source row stride stride0 (elements).  D# built per ISA ch.8.
__device__ __forceinline__ void tdm_load_2d(
    const __bf16* gsrc, unsigned lds_byte_off,
    unsigned tensor_d0, unsigned tensor_d1, unsigned long long stride0,
    unsigned tile_d0, unsigned tile_d1) {
  unsigned long long ga = (unsigned long long)(const void*)gsrc;
  tdm_u32x4 g0;
  g0[0] = 1u;                                      // count=1, user mode
  g0[1] = lds_byte_off;                            // lds_addr
  g0[2] = (unsigned)ga;                            // global_addr[31:0]
  g0[3] = (unsigned)((ga >> 32) & 0x1FFFFFFu) | 0x80000000u;  // [56:32] | type=2
  tdm_i32x8 g1;
  g1[0] = (int)(1u << 16);                         // data_size=1 (2 bytes)
  g1[1] = (int)((tensor_d0 & 0xFFFFu) << 16);      // tensor_dim0 lo16
  g1[2] = (int)((tensor_d0 >> 16) | ((tensor_d1 & 0xFFFFu) << 16));
  g1[3] = (int)(((tensor_d1 >> 16) & 0xFFFFu) | (tile_d0 << 16));
  g1[4] = (int)tile_d1;                            // tile_dim1 | tile_dim2=0
  g1[5] = (int)(unsigned)(stride0 & 0xFFFFFFFFull);
  g1[6] = (int)(unsigned)((stride0 >> 32) & 0xFFFFull);
  g1[7] = 0;
  tdm_i32x4 z4 = {0, 0, 0, 0};
#if __clang_major__ >= 23
  tdm_i32x8 z8 = {0, 0, 0, 0, 0, 0, 0, 0};
  __builtin_amdgcn_tensor_load_to_lds(g0, g1, z4, z4, z8, 0);
#else
  __builtin_amdgcn_tensor_load_to_lds(g0, g1, z4, z4, 0);
#endif
}
__device__ __forceinline__ unsigned lds_off(const void* p) {
  return (unsigned)(unsigned long long)p;          // flat LDS addr[31:0] = LDS byte offset
}
#endif

// ---------- stage 0: fp32 -> bf16 convert ----------

__global__ void cvt_f32_bf16(const float* __restrict__ src,
                             __bf16* __restrict__ dst, int n) {
  int i = blockIdx.x * blockDim.x + threadIdx.x;
  int stride = gridDim.x * blockDim.x;
  for (; i < n; i += stride) dst[i] = f2bf(src[i]);
}

// ---------- LDS-tiled GEMM core (128M x 128N block, 8 waves) ----------
// Computes acc[8] (wave strip: rows waveId*16..+16, all 128 N) with
// double-buffered TDM panel staging.  Abase/Bbase: block panel starts,
// row stride D_MODEL, K contiguous.

__device__ __forceinline__ void gemm_block_128x128(
    const __bf16* Abase, const __bf16* Bbase,
    __bf16 (*sA)[128 * 32], __bf16 (*sB)[128 * 32],
    int wave, int lane, v8f acc[8]) {
  const int tid = wave * 32 + lane;
  const int laneHalf = lane >> 4, ln = lane & 15;
#pragma unroll
  for (int t = 0; t < 8; ++t)
#pragma unroll
    for (int r = 0; r < 8; ++r) acc[t][r] = 0.f;

#if HAVE_TDM
  if (wave == 0) {   // prologue: stage k-step 0 into buffer 0
    tdm_load_2d(Abase, lds_off(&sA[0][0]), D_MODEL, 8192, D_MODEL, 32, 128);
    tdm_load_2d(Bbase, lds_off(&sB[0][0]), D_MODEL, 8192, D_MODEL, 32, 128);
  }
#endif
  for (int step = 0; step < D_MODEL / 32; ++step) {
#if HAVE_TDM
    const int buf = step & 1;
    if (wave == 0) {
      if (step + 1 < D_MODEL / 32) {   // prefetch next panel pair
        tdm_load_2d(Abase + (step + 1) * 32, lds_off(&sA[buf ^ 1][0]),
                    D_MODEL, 8192, D_MODEL, 32, 128);
        tdm_load_2d(Bbase + (step + 1) * 32, lds_off(&sB[buf ^ 1][0]),
                    D_MODEL, 8192, D_MODEL, 32, 128);
        __builtin_amdgcn_s_wait_tensorcnt(2);   // current pair complete
      } else {
        __builtin_amdgcn_s_wait_tensorcnt(0);
      }
    }
    __syncthreads();
#else
    const int buf = 0;
    {   // cooperative staging: 256 threads x 32B each per panel
      const int row = tid >> 1, half = tid & 1;
      *(v16bf*)&sA[0][row * 32 + half * 16] =
          *(const v16bf*)&Abase[(size_t)row * D_MODEL + step * 32 + half * 16];
      *(v16bf*)&sB[0][row * 32 + half * 16] =
          *(const v16bf*)&Bbase[(size_t)row * D_MODEL + step * 32 + half * 16];
    }
    __syncthreads();
#endif
    const v16bf a = load_a_frag(&sA[buf][(wave * 16 + ln) * 32], laneHalf);
#pragma unroll
    for (int t = 0; t < 8; ++t) {
      const v16bf b = load_b_frag(&sB[buf][(t * 16 + ln) * 32], laneHalf);
      acc[t] = wmma_bf16(a, b, acc[t]);
    }
    __syncthreads();   // seal reads before the next prefetch overwrites
  }
}

// ---------- stage 1: fused QKV projection ----------
// Q,K stored (B,H,S,Dk) bf16; V stored transposed (B,H,Dk,S) bf16.

__global__ __launch_bounds__(256) void qkv_kernel(
    const __bf16* __restrict__ xb,
    const __bf16* __restrict__ wq, const __bf16* __restrict__ wk,
    const __bf16* __restrict__ wv,
    const float* __restrict__ bq, const float* __restrict__ bk,
    const float* __restrict__ bv,
    __bf16* __restrict__ qm, __bf16* __restrict__ km,
    __bf16* __restrict__ vtm)
{
  __shared__ __bf16 sA[2][128 * 32];
  __shared__ __bf16 sB[2][128 * 32];
  const int wave = threadIdx.x >> 5, lane = threadIdx.x & 31;
  const int laneHalf = lane >> 4, ln = lane & 15;
  const int proj = blockIdx.x >> 9;             // 512 blocks per projection
  const int rem = blockIdx.x & 511;
  const int mb = rem >> 3, nb = rem & 7;        // 64 M-blocks x 8 N-blocks
  const __bf16* wb  = proj == 0 ? wq : (proj == 1 ? wk : wv);
  const float* bias = proj == 0 ? bq : (proj == 1 ? bk : bv);

  v8f acc[8];
  gemm_block_128x128(xb + (size_t)mb * 128 * D_MODEL,
                     wb + (size_t)nb * 128 * D_MODEL, sA, sB, wave, lane, acc);

#pragma unroll
  for (int t = 0; t < 8; ++t) {
    const int e = nb * 128 + t * 16 + ln;
    const float bval = bias[e];
    const int h = e >> 6, dk = e & 63;
#pragma unroll
    for (int r = 0; r < 8; ++r) {
      const int grow = mb * 128 + wave * 16 + r + 8 * laneHalf;   // b*SEQ + s
      const int bb = grow >> 11, s = grow & (SEQ - 1);
      const __bf16 val = f2bf(acc[t][r] + bval);
      const size_t bh = (size_t)(bb * N_HEADS + h);
      if (proj == 2)      vtm[(bh * DK + dk) * SEQ + s] = val;   // (B,H,Dk,S)
      else if (proj == 1) km [(bh * SEQ + s) * DK + dk] = val;   // (B,H,S,Dk)
      else                qm [(bh * SEQ + s) * DK + dk] = val;
    }
  }
}

// ---------- stage 2: flash attention ----------
// One wave per 16-query tile; all 8 waves of a block share one (b,h) and the
// TDM-staged K/V tiles.

__global__ __launch_bounds__(256) void attn_kernel(
    const __bf16* __restrict__ qm, const __bf16* __restrict__ km,
    const __bf16* __restrict__ vtm, const int* __restrict__ amask,
    __bf16* __restrict__ ao)
{
  __shared__ __bf16 sK[2][32 * 64];    // keys x dk
  __shared__ __bf16 sV[2][64 * 32];    // dk x keys (from transposed V)
  __shared__ __bf16 pbuf[8][16][40];   // per-wave 16x32 P tile, padded rows
  const int tid = threadIdx.x;
  const int wave = tid >> 5, lane = tid & 31;
  const int laneHalf = lane >> 4, ln = lane & 15;
  const int bh = blockIdx.x >> 4;                 // 64 (b,h) pairs
  const int qt = (blockIdx.x & 15) * 8 + wave;    // 128 q-tiles
  const int b = bh >> 4, h = bh & 15;

  const __bf16* qbase = qm  + ((size_t)bh * SEQ + qt * 16) * DK;
  const __bf16* kbase = km  + (size_t)bh * SEQ * DK;
  const __bf16* vbase = vtm + (size_t)bh * DK * SEQ;
  const int* mbase = amask + b * SEQ;

  // Q fragments for dk 0..31 / 32..63, reused across all key tiles
  const v16bf qa0 = load_a_frag(qbase + (size_t)ln * DK, laneHalf);
  const v16bf qa1 = load_a_frag(qbase + (size_t)ln * DK + 32, laneHalf);

  float mrow[8], lrow[8];
  v8f o[4];
#pragma unroll
  for (int r = 0; r < 8; ++r) { mrow[r] = -3.0e38f; lrow[r] = 0.f; }
#pragma unroll
  for (int t = 0; t < 4; ++t)
#pragma unroll
    for (int r = 0; r < 8; ++r) o[t][r] = 0.f;

#if HAVE_TDM
  if (wave == 0) {   // prologue: stage key-step 0
    tdm_load_2d(kbase, lds_off(&sK[0][0]), DK, SEQ, DK, DK, 32);
    tdm_load_2d(vbase, lds_off(&sV[0][0]), SEQ, DK, SEQ, 32, DK);
  }
#endif
  for (int step = 0; step < SEQ / 32; ++step) {
    const int kb = step * 32;
#if HAVE_TDM
    const int buf = step & 1;
    if (wave == 0) {
      if (step + 1 < SEQ / 32) {
        tdm_load_2d(kbase + (size_t)(kb + 32) * DK, lds_off(&sK[buf ^ 1][0]),
                    DK, SEQ, DK, DK, 32);
        tdm_load_2d(vbase + kb + 32, lds_off(&sV[buf ^ 1][0]),
                    SEQ, DK, SEQ, 32, DK);
        __builtin_amdgcn_s_wait_tensorcnt(2);
      } else {
        __builtin_amdgcn_s_wait_tensorcnt(0);
      }
    }
    __syncthreads();
#else
    const int buf = 0;
    if (tid < 128) {                       // K tile: 32 rows x 64
      const int row = tid >> 2, q = tid & 3;
      *(v16bf*)&sK[0][row * 64 + q * 16] =
          *(const v16bf*)&kbase[(size_t)(kb + row) * DK + q * 16];
    } else {                               // V tile: 64 rows x 32
      const int t2 = tid - 128;
      const int row = t2 >> 1, hf = t2 & 1;
      *(v16bf*)&sV[0][row * 32 + hf * 16] =
          *(const v16bf*)&vbase[(size_t)row * SEQ + kb + hf * 16];
    }
    __syncthreads();
#endif
    // --- scores: two 16x16 tiles over keys [kb,kb+16) and [kb+16,kb+32) ---
    v8f s0, s1;
#pragma unroll
    for (int r = 0; r < 8; ++r) { s0[r] = 0.f; s1[r] = 0.f; }
    v16bf kf;
    kf = load_b_frag(&sK[buf][ln * 64], laneHalf);            s0 = wmma_bf16(qa0, kf, s0);
    kf = load_b_frag(&sK[buf][ln * 64 + 32], laneHalf);       s0 = wmma_bf16(qa1, kf, s0);
    kf = load_b_frag(&sK[buf][(16 + ln) * 64], laneHalf);     s1 = wmma_bf16(qa0, kf, s1);
    kf = load_b_frag(&sK[buf][(16 + ln) * 64 + 32], laneHalf);s1 = wmma_bf16(qa1, kf, s1);

    const float mk0 = (1.f - (float)mbase[kb + ln])      * -10000.f;
    const float mk1 = (1.f - (float)mbase[kb + 16 + ln]) * -10000.f;

    // --- online softmax (rows live per half-wave: M = r + 8*laneHalf) ---
    float alpha[8];
#pragma unroll
    for (int r = 0; r < 8; ++r) {
      float x0 = s0[r] * 0.125f + mk0;   // 1/sqrt(64)
      float x1 = s1[r] * 0.125f + mk1;
      float mx = fmaxf(x0, x1);
#pragma unroll
      for (int d = 1; d < 16; d <<= 1) mx = fmaxf(mx, __shfl_xor(mx, d, 32));
      const float mn = fmaxf(mrow[r], mx);
      alpha[r] = __expf(mrow[r] - mn);
      mrow[r] = mn;
      const float p0 = __expf(x0 - mn);
      const float p1 = __expf(x1 - mn);
      float rs = p0 + p1;
#pragma unroll
      for (int d = 1; d < 16; d <<= 1) rs += __shfl_xor(rs, d, 32);
      lrow[r] = lrow[r] * alpha[r] + rs;
      const int rowm = r + 8 * laneHalf;
      pbuf[wave][rowm][ln]      = f2bf(p0);   // C-layout -> LDS (row, key)
      pbuf[wave][rowm][16 + ln] = f2bf(p1);
    }
    __syncthreads();
    const v16bf pa = load_a_frag(&pbuf[wave][ln][0], laneHalf);   // A-layout
    __syncthreads();

    // --- O += P * V  (B-fragments from staged transposed V) ---
#pragma unroll
    for (int t = 0; t < 4; ++t) {
#pragma unroll
      for (int r = 0; r < 8; ++r) o[t][r] *= alpha[r];
      const v16bf vf = load_b_frag(&sV[buf][(t * 16 + ln) * 32], laneHalf);
      o[t] = wmma_bf16(pa, vf, o[t]);
    }
    __syncthreads();   // seal tile reads before next prefetch/stage
  }

  // --- normalize and write attn output in (B,S,D) bf16 ---
#pragma unroll
  for (int r = 0; r < 8; ++r) {
    const float inv = lrow[r] > 0.f ? 1.f / lrow[r] : 0.f;
    const int srow = qt * 16 + r + 8 * laneHalf;
#pragma unroll
    for (int t = 0; t < 4; ++t) {
      const int col = h * DK + t * 16 + ln;
      ao[((size_t)b * SEQ + srow) * D_MODEL + col] = f2bf(o[t][r] * inv);
    }
  }
}

// ---------- stage 3: output projection (fp32 result) ----------

__global__ __launch_bounds__(256) void out_kernel(
    const __bf16* __restrict__ ao, const __bf16* __restrict__ wo,
    const float* __restrict__ bo, float* __restrict__ out)
{
  __shared__ __bf16 sA[2][128 * 32];
  __shared__ __bf16 sB[2][128 * 32];
  const int wave = threadIdx.x >> 5, lane = threadIdx.x & 31;
  const int laneHalf = lane >> 4, ln = lane & 15;
  const int mb = blockIdx.x >> 3, nb = blockIdx.x & 7;

  v8f acc[8];
  gemm_block_128x128(ao + (size_t)mb * 128 * D_MODEL,
                     wo + (size_t)nb * 128 * D_MODEL, sA, sB, wave, lane, acc);

#pragma unroll
  for (int t = 0; t < 8; ++t) {
    const int e = nb * 128 + t * 16 + ln;
    const float bval = bo[e];
#pragma unroll
    for (int r = 0; r < 8; ++r) {
      const int grow = mb * 128 + wave * 16 + r + 8 * laneHalf;
      out[(size_t)grow * D_MODEL + e] = acc[t][r] + bval;
    }
  }
}

// ---------- host launcher ----------

extern "C" void kernel_launch(void* const* d_in, const int* in_sizes, int n_in,
                              void* d_out, int out_size, void* d_ws, size_t ws_size,
                              hipStream_t stream) {
  (void)in_sizes; (void)n_in; (void)out_size; (void)ws_size;
  const float* x    = (const float*)d_in[0];
  const int*  amask = (const int*)d_in[1];
  const float* w_q  = (const float*)d_in[2];
  const float* b_q  = (const float*)d_in[3];
  const float* w_k  = (const float*)d_in[4];
  const float* b_k  = (const float*)d_in[5];
  const float* w_v  = (const float*)d_in[6];
  const float* b_v  = (const float*)d_in[7];
  const float* w_o  = (const float*)d_in[8];
  const float* b_o  = (const float*)d_in[9];
  float* out = (float*)d_out;

  char* ws = (char*)d_ws;
  size_t off = 0;
  auto take = [&](size_t bytes) {
    void* p = ws + off;
    off += (bytes + 255) & ~(size_t)255;
    return p;
  };
  const size_t NX = (size_t)MROWS * D_MODEL;
  const size_t NW = (size_t)D_MODEL * D_MODEL;
  const size_t NH = (size_t)BATCH * N_HEADS * SEQ * DK;
  __bf16* xb  = (__bf16*)take(NX * 2);
  __bf16* wqb = (__bf16*)take(NW * 2);
  __bf16* wkb = (__bf16*)take(NW * 2);
  __bf16* wvb = (__bf16*)take(NW * 2);
  __bf16* wob = (__bf16*)take(NW * 2);
  __bf16* qm  = (__bf16*)take(NH * 2);
  __bf16* km  = (__bf16*)take(NH * 2);
  __bf16* vtm = (__bf16*)take(NH * 2);
  __bf16* ao  = (__bf16*)take(NX * 2);

  // stage 0: converts
  cvt_f32_bf16<<<8192, 256, 0, stream>>>(x, xb, (int)NX);
  cvt_f32_bf16<<<4096, 256, 0, stream>>>(w_q, wqb, (int)NW);
  cvt_f32_bf16<<<4096, 256, 0, stream>>>(w_k, wkb, (int)NW);
  cvt_f32_bf16<<<4096, 256, 0, stream>>>(w_v, wvb, (int)NW);
  cvt_f32_bf16<<<4096, 256, 0, stream>>>(w_o, wob, (int)NW);

  // stage 1: QKV projections — 3 proj x 64 M-blocks x 8 N-blocks
  qkv_kernel<<<1536, 256, 0, stream>>>(xb, wqb, wkb, wvb, b_q, b_k, b_v,
                                       qm, km, vtm);

  // stage 2: flash attention — 64 (b,h) x 16 blocks of 8 q-tiles
  attn_kernel<<<1024, 256, 0, stream>>>(qm, km, vtm, amask, ao);

  // stage 3: output projection — 64 M-blocks x 8 N-blocks
  out_kernel<<<512, 256, 0, stream>>>(ao, wob, b_o, out);
}
